// Transformer_57569741636181
// MI455X (gfx1250) — matrix operand order repaired and checked by reading, not compiled, not run on hardware
//
#include <hip/hip_runtime.h>

typedef __attribute__((ext_vector_type(16))) _Float16 v16h;
typedef __attribute__((ext_vector_type(8)))  _Float16 v8h;
typedef __attribute__((ext_vector_type(8)))  float    v8f;
typedef __attribute__((ext_vector_type(4)))  unsigned int u32x4;
typedef __attribute__((ext_vector_type(8)))  int          i32x8;
typedef __attribute__((ext_vector_type(4)))  int          i32x4;

#define B_      16
#define C_      512
#define N_      784      // 28*28 tokens
#define NP_     800      // padded to multiple of 32 for WMMA K-loop
#define HEADS_  8
#define DHEAD_  32
#define INNER_  256
#define HIDDEN_ 2048
#define BN_     (B_ * N_)        // 12544
#define BH_     (B_ * HEADS_)    // 128
#define NT_     49               // 784/16 tiles

// LDS staging: padded row stride 80B (64B data + 16B TDM pad) -> conflict-free ds_load_b128
#define LDSROW_     80
#define A_PANEL_SZ  (64 * LDSROW_)    // 5120 B
#define B_PANEL_SZ  (128 * LDSROW_)   // 10240 B
#define BUF_SZ      (A_PANEL_SZ + B_PANEL_SZ)

// ---------------- WMMA helpers (gfx1250 wave32) ----------------

__device__ __forceinline__ v8f wmma_f16(v16h a, v16h b, v8f c) {
    return __builtin_amdgcn_wmma_f32_16x16x32_f16(
        /*neg_a=*/false, a, /*neg_b=*/false, b,
        /*c_mod=*/(short)0, c, /*reuse_a=*/false, /*reuse_b=*/false);
}

__device__ __forceinline__ v16h cat8(v8h lo, v8h hi) {
    return __builtin_shufflevector(lo, hi, 0,1,2,3,4,5,6,7,8,9,10,11,12,13,14,15);
}

// A-matrix 16x32 f16 tile, row-major with leading dim ld (global memory path).
// Lane l holds row m=l&15; halves 0..7 = K {8*hi + 0..7}, halves 8..15 = K {16 + 8*hi + 0..7}.
__device__ __forceinline__ v16h load_a_tile(const _Float16* __restrict__ base, int ld) {
    const int l   = threadIdx.x & 31;
    const int row = l & 15;
    const int k0  = (l >> 4) << 3;              // 0 or 8
    const _Float16* p = base + (size_t)row * ld + k0;
    return cat8(*reinterpret_cast<const v8h*>(p), *reinterpret_cast<const v8h*>(p + 16));
}

// B-matrix 32x16 f16 tile from Bt (B transposed, row-major [N][K], leading dim ld).
// Lane l holds col n=l&15; halves 0..7 = K {8*hi + 0..7}, halves 8..15 = K {16 + 8*hi + 0..7}.
__device__ __forceinline__ v16h load_b_tile(const _Float16* __restrict__ baseT, int ld) {
    const int l   = threadIdx.x & 31;
    const int col = l & 15;
    const int k0  = (l >> 4) << 3;              // 0 or 8
    const _Float16* p = baseT + (size_t)col * ld + k0;
    return cat8(*reinterpret_cast<const v8h*>(p), *reinterpret_cast<const v8h*>(p + 16));
}

// Same tiles out of the TDM-padded LDS panels (row stride LDSROW_ bytes).
__device__ __forceinline__ v16h lds_a_tile(const char* base, int row0) {
    const int l = threadIdx.x & 31;
    const _Float16* p = (const _Float16*)(base + (size_t)(row0 + (l & 15)) * LDSROW_)
                        + ((l >> 4) << 3);
    return cat8(*reinterpret_cast<const v8h*>(p), *reinterpret_cast<const v8h*>(p + 16));
}
__device__ __forceinline__ v16h lds_b_tile(const char* base, int row0) {
    const int l = threadIdx.x & 31;
    const _Float16* p = (const _Float16*)(base + (size_t)(row0 + (l & 15)) * LDSROW_)
                        + ((l >> 4) << 3);
    return cat8(*reinterpret_cast<const v8h*>(p), *reinterpret_cast<const v8h*>(p + 16));
}

__device__ __forceinline__ float gelu_exact(float x) {
    return 0.5f * x * (1.0f + erff(x * 0.70710678118654752f));
}

// ---------------- Tensor Data Mover: 2-D f16 tile -> LDS (ISA 08 / D# groups 0-1) ----------------
// tile_d0 elements per row (contiguous in memory), tile_d1 rows, row stride d0_stride elems.
// pad: insert 4 DWORDs (16B) after every 16 DWORDs (64B row) -> LDS row stride 80B.
__device__ __forceinline__ void tdm_load_2d_f16(
    unsigned lds_addr, const void* gptr,
    unsigned tensor_d0, unsigned tensor_d1, unsigned d0_stride,
    unsigned tile_d0, unsigned tile_d1)
{
    const unsigned long long ga = (unsigned long long)(uintptr_t)gptr;
    u32x4 g0;
    g0[0] = 1u;                                              // count=1, no gather
    g0[1] = lds_addr;                                        // LDS byte address
    g0[2] = (unsigned)(ga & 0xFFFFFFFFull);                  // global_addr[31:0]
    g0[3] = (unsigned)((ga >> 32) & 0x01FFFFFFull)           // global_addr[56:32]
          | (2u << 30);                                      // type = 2 (image)
    i32x8 g1;
    g1[0] = (int)((1u << 16)        // data_size = 1 -> 2 bytes
                | (1u << 20)        // pad_enable
                | (3u << 22)        // pad_interval code 3 -> every 16 DWORDs
                | (3u << 25));      // pad_amount  code 3 -> 4 DWORDs
    g1[1] = (int)((tensor_d0 & 0xFFFFu) << 16);                               // dim0[15:0]
    g1[2] = (int)(((tensor_d0 >> 16) & 0xFFFFu) | ((tensor_d1 & 0xFFFFu) << 16));
    g1[3] = (int)(((tensor_d1 >> 16) & 0xFFFFu) | ((tile_d0 & 0xFFFFu) << 16));
    g1[4] = (int)(tile_d1 & 0xFFFFu);                                         // tile_dim1, tile_dim2=0
    g1[5] = (int)d0_stride;                                                   // dim0_stride[31:0]
    g1[6] = 0;                                                                // stride[47:32], dim1_stride lo
    g1[7] = 0;
    const i32x4 z4 = {0, 0, 0, 0};                           // groups 2/3 disabled (2-D tile)
    const i32x8 z8 = {0, 0, 0, 0, 0, 0, 0, 0};
    __builtin_amdgcn_tensor_load_to_lds(g0, g1, z4, z4, z8, 0);   // 6-arg form (clang-23 lane)
}

// ---------------- Generic WMMA GEMM with TDM double-buffered LDS staging ----------------
// Out = act(A[M,K] @ B[K,N] + bias), B supplied transposed as Bt[N,K].
// mode 0: O16 = result   mode 1: O16 = gelu(result)   mode 2: O32 = Res + result
// Block 256 = 8 waves (2 M x 4 N), block tile 64x128, wave tile 32x32, K step 32.
__global__ __launch_bounds__(256)
void gemm16_kernel(const _Float16* __restrict__ A, const _Float16* __restrict__ Bt,
                   const float* __restrict__ bias,
                   _Float16* __restrict__ O16,
                   float* __restrict__ O32, const float* __restrict__ Res,
                   int M, int Nn, int K, int mode)
{
    __shared__ alignas(16) char lds_mem[2 * BUF_SZ];
    const unsigned lds_base = (unsigned)(uintptr_t)&lds_mem[0];   // LDS aperture: addr[31:0]

    const int wave = threadIdx.x >> 5;
    const size_t mblock = (size_t)blockIdx.y * 64;
    const size_t nblock = (size_t)blockIdx.x * 128;
    const int steps = K >> 5;

    // One wave drives the TDM; EXEC is ignored by tensor ops, TENSORcnt is per-wave.
    if (threadIdx.x < 32) {
        tdm_load_2d_f16(lds_base,              A  + mblock * K, K, (unsigned)M,  K, 32, 64);
        tdm_load_2d_f16(lds_base + A_PANEL_SZ, Bt + nblock * K, K, (unsigned)Nn, K, 32, 128);
    }

    v8f acc00 = {}, acc01 = {}, acc10 = {}, acc11 = {};
    const int mw = (wave & 1) * 32;
    const int nw = (wave >> 1) * 32;

    for (int i = 0; i < steps; ++i) {
        if (threadIdx.x < 32) {
            if (i + 1 < steps) {
                const unsigned nb = lds_base + ((i + 1) & 1) * BUF_SZ;
                const int kn = (i + 1) << 5;
                tdm_load_2d_f16(nb,              A  + mblock * K + kn, K, (unsigned)M,  K, 32, 64);
                tdm_load_2d_f16(nb + A_PANEL_SZ, Bt + nblock * K + kn, K, (unsigned)Nn, K, 32, 128);
                __builtin_amdgcn_s_wait_tensorcnt((short)2);   // current panel done (in-order)
            } else {
                __builtin_amdgcn_s_wait_tensorcnt((short)0);
            }
        }
        __syncthreads();                                       // panel visible to all waves

        const char* abase = lds_mem + (i & 1) * BUF_SZ;
        const char* bbase = abase + A_PANEL_SZ;
        v16h a0 = lds_a_tile(abase, mw);
        v16h a1 = lds_a_tile(abase, mw + 16);
        v16h b0 = lds_b_tile(bbase, nw);
        v16h b1 = lds_b_tile(bbase, nw + 16);
        acc00 = wmma_f16(a0, b0, acc00);
        acc01 = wmma_f16(a0, b1, acc01);
        acc10 = wmma_f16(a1, b0, acc10);
        acc11 = wmma_f16(a1, b1, acc11);

        __syncthreads();                                       // reads done before buffer reuse
    }

    const int l  = threadIdx.x & 31;
    const int nl = l & 15;
    const int ml = (l >> 4) * 8;
    for (int mt = 0; mt < 2; ++mt) {
        for (int nt = 0; nt < 2; ++nt) {
            v8f acc = (mt == 0) ? ((nt == 0) ? acc00 : acc01)
                                : ((nt == 0) ? acc10 : acc11);
            const size_t col = nblock + (size_t)nw + (size_t)nt * 16 + nl;
            const float bs = bias ? bias[col] : 0.0f;
            const size_t row0 = mblock + (size_t)mw + (size_t)mt * 16 + ml;
            for (int r = 0; r < 8; ++r) {
                float v = acc[r] + bs;
                if (mode == 1) v = gelu_exact(v);
                const size_t idx = (row0 + r) * (size_t)Nn + col;
                if (mode == 2) O32[idx] = Res[idx] + v;
                else           O16[idx] = (_Float16)v;
            }
        }
    }
}

// ---------------- LayerNorm (optionally transposed input read) ----------------
// element (b,n,c) at in[b*C*N + n*sn + c*sc]; LN1: sn=1,sc=N (reads x[B,C,N]); LN2: sn=C,sc=1.
__global__ __launch_bounds__(256)
void ln_kernel(const float* __restrict__ in, const float* __restrict__ g,
               const float* __restrict__ bt, float* __restrict__ raw,
               _Float16* __restrict__ h16, int sn, int sc)
{
    __shared__ float s1[256];
    __shared__ float s2[256];
    const int token = blockIdx.x;                 // 0 .. B*N-1
    const int b = token / N_, n = token - b * N_;
    const float* base = in + (size_t)b * C_ * N_ + (size_t)n * sn;
    const int t = threadIdx.x;
    const float v0 = base[(size_t)t * sc];
    const float v1 = base[(size_t)(t + 256) * sc];
    s1[t] = v0 + v1;
    s2[t] = v0 * v0 + v1 * v1;
    __syncthreads();
    for (int off = 128; off > 0; off >>= 1) {
        if (t < off) { s1[t] += s1[t + off]; s2[t] += s2[t + off]; }
        __syncthreads();
    }
    const float mean = s1[0] * (1.0f / C_);
    const float var  = s2[0] * (1.0f / C_) - mean * mean;
    const float rs   = rsqrtf(var + 1e-5f);
    const size_t ob  = (size_t)token * C_;
    h16[ob + t]       = (_Float16)((v0 - mean) * rs * g[t]       + bt[t]);
    h16[ob + t + 256] = (_Float16)((v1 - mean) * rs * g[t + 256] + bt[t + 256]);
    if (raw) { raw[ob + t] = v0; raw[ob + t + 256] = v1; }
}

// ---------------- Weight convert + transpose: w[K,N] f32 -> wt[N,K] f16 ----------------
__global__ __launch_bounds__(256)
void wconv_kernel(const float* __restrict__ w, _Float16* __restrict__ wt, int K, int Nn)
{
    const size_t idx = (size_t)blockIdx.x * 256 + threadIdx.x;
    if (idx >= (size_t)K * Nn) return;
    const int k = (int)(idx % K);
    const int n = (int)(idx / K);
    wt[idx] = (_Float16)w[(size_t)k * Nn + n];
}

// ---------------- Repack QKV: [BN,768] -> q(scaled)/k [BH,784,32], vT [BH,32,800] padded ----------------
__global__ __launch_bounds__(256)
void repack_qkv_kernel(const _Float16* __restrict__ qkv,
                       _Float16* __restrict__ q, _Float16* __restrict__ k,
                       _Float16* __restrict__ vT)
{
    const size_t idx = (size_t)blockIdx.x * 256 + threadIdx.x;
    const size_t total = (size_t)BH_ * NP_ * DHEAD_;
    if (idx >= total) return;
    const int d  = (int)(idx & 31);
    const int n  = (int)((idx >> 5) % NP_);
    const int bh = (int)(idx / ((size_t)NP_ * DHEAD_));
    const int b = bh >> 3, h = bh & 7;
    if (n < N_) {
        const _Float16* src = qkv + ((size_t)b * N_ + n) * (3 * INNER_) + h * DHEAD_ + d;
        q[((size_t)bh * N_ + n) * DHEAD_ + d] =
            (_Float16)((float)src[0] * 0.17677669529663687f);     // 1/sqrt(32)
        k[((size_t)bh * N_ + n) * DHEAD_ + d] = src[INNER_];
        vT[((size_t)bh * DHEAD_ + d) * NP_ + n] = src[2 * INNER_];
    } else {
        vT[((size_t)bh * DHEAD_ + d) * NP_ + n] = (_Float16)0.0f;
    }
}

// ---------------- Attention scores: S[bh,i,j] = q.k + relbias (single K=32 WMMA per 16x16 tile) ----------------
__global__ __launch_bounds__(256)
void attn_scores_kernel(const _Float16* __restrict__ q, const _Float16* __restrict__ kk,
                        const float* __restrict__ rel, float* __restrict__ S)
{
    const int bh = blockIdx.y;
    const int h  = bh & 7;
    const int tile = blockIdx.x * 8 + (threadIdx.x >> 5);
    if (tile >= NT_ * NT_) return;               // wave-uniform exit: EXEC stays all-1 at WMMA
    const int mt = tile / NT_, nt = tile - mt * NT_;

    const _Float16* qb = q  + (size_t)bh * N_ * DHEAD_ + (size_t)mt * 16 * DHEAD_;
    const _Float16* kb = kk + (size_t)bh * N_ * DHEAD_ + (size_t)nt * 16 * DHEAD_;
    v16h a = load_a_tile(qb, DHEAD_);
    v16h b = load_b_tile(kb, DHEAD_);            // Bt[j][d] = K[j][d] directly
    v8f  c = {};
    c = wmma_f16(a, b, c);

    const int l  = threadIdx.x & 31;
    const int nl = l & 15, ml = (l >> 4) * 8;
    const int tj = nt * 16 + nl;
    const int j0 = tj / 28, j1 = tj - j0 * 28;
    float* srow = S + (size_t)bh * N_ * NP_;
    for (int r = 0; r < 8; ++r) {
        const int ti = mt * 16 + ml + r;
        const int i0 = ti / 28, i1 = ti - i0 * 28;
        const int ridx = (i0 - j0 + 27) * 55 + (i1 - j1 + 27);
        srow[(size_t)ti * NP_ + tj] = c[r] + rel[ridx * HEADS_ + h];
    }
}

// ---------------- Row softmax (in-place alias safe: row staged in registers), pads cols 784..799 ----------------
__global__ __launch_bounds__(256)
void softmax_kernel(const float* __restrict__ S, _Float16* __restrict__ P)
{
    __shared__ float red[256];
    const size_t row = blockIdx.x;
    const float* s = S + row * NP_;
    _Float16*    p = P + row * NP_;
    const int t = threadIdx.x;

    float vals[4];
    float lm = -3.0e38f;
    for (int i = 0; i < 4; ++i) {
        const int j = t + i * 256;
        vals[i] = (j < N_) ? s[j] : -3.0e38f;
        lm = fmaxf(lm, vals[i]);
    }
    red[t] = lm; __syncthreads();
    for (int off = 128; off > 0; off >>= 1) {
        if (t < off) red[t] = fmaxf(red[t], red[t + off]);
        __syncthreads();
    }
    const float m = red[0];
    __syncthreads();

    float ev[4];
    float ls = 0.0f;
    for (int i = 0; i < 4; ++i) {
        const int j = t + i * 256;
        ev[i] = (j < N_) ? expf(vals[i] - m) : 0.0f;
        ls += ev[i];
    }
    red[t] = ls; __syncthreads();
    for (int off = 128; off > 0; off >>= 1) {
        if (t < off) red[t] += red[t + off];
        __syncthreads();
    }
    const float inv = 1.0f / red[0];
    for (int i = 0; i < 4; ++i) {
        const int j = t + i * 256;
        if (j < N_) p[j] = (_Float16)(ev[i] * inv);
    }
    if (t < NP_ - N_) p[N_ + t] = (_Float16)0.0f;    // zero padding for the K=800 GEMM
}

// ---------------- Attention output: O[bh,i,d] = P[784,800] @ V[800,32]; wave = one 16x32 tile ----------------
__global__ __launch_bounds__(256)
void attn_out_kernel(const _Float16* __restrict__ P, const _Float16* __restrict__ vT,
                     _Float16* __restrict__ o16)
{
    const int bh = blockIdx.y;
    const int b = bh >> 3, h = bh & 7;
    const int mt = blockIdx.x * 8 + (threadIdx.x >> 5);
    if (mt >= NT_) return;                       // wave-uniform exit
    const _Float16* Pb = P  + (size_t)bh * N_ * NP_ + (size_t)mt * 16 * NP_;
    const _Float16* Vb = vT + (size_t)bh * DHEAD_ * NP_;

    v8f acc0 = {}, acc1 = {};
    for (int k = 0; k < NP_; k += 32) {
        v16h a  = load_a_tile(Pb + k, NP_);
        v16h b0 = load_b_tile(Vb + k, NP_);
        v16h b1 = load_b_tile(Vb + (size_t)16 * NP_ + k, NP_);
        acc0 = wmma_f16(a, b0, acc0);
        acc1 = wmma_f16(a, b1, acc1);
    }
    const int l  = threadIdx.x & 31;
    const int nl = l & 15, ml = (l >> 4) * 8;
    _Float16* ob = o16 + ((size_t)b * N_ + (size_t)mt * 16) * INNER_ + h * DHEAD_;
    for (int r = 0; r < 8; ++r) {
        ob[(size_t)(ml + r) * INNER_ + nl]      = (_Float16)acc0[r];
        ob[(size_t)(ml + r) * INNER_ + 16 + nl] = (_Float16)acc1[r];
    }
}

// ---------------- Final layout: xs3[B,N,C] -> out[B,C,28,28] ----------------
__global__ __launch_bounds__(256)
void out_transpose_kernel(const float* __restrict__ xs3, float* __restrict__ out)
{
    const size_t idx = (size_t)blockIdx.x * 256 + threadIdx.x;
    if (idx >= (size_t)B_ * C_ * N_) return;
    const int n = (int)(idx % N_);
    const int c = (int)((idx / N_) % C_);
    const int b = (int)(idx / ((size_t)N_ * C_));
    out[idx] = xs3[((size_t)b * N_ + n) * C_ + c];
}

// ---------------- Host orchestration ----------------
static inline size_t align256(size_t x) { return (x + 255) & ~(size_t)255; }

extern "C" void kernel_launch(void* const* d_in, const int* in_sizes, int n_in,
                              void* d_out, int out_size, void* d_ws, size_t ws_size,
                              hipStream_t stream) {
    (void)in_sizes; (void)n_in; (void)out_size; (void)ws_size;
    const float* x      = (const float*)d_in[0];
    const float* ln1_g  = (const float*)d_in[1];
    const float* ln1_b  = (const float*)d_in[2];
    const float* qkv_w  = (const float*)d_in[3];
    const float* qkv_b  = (const float*)d_in[4];
    const float* out_w  = (const float*)d_in[5];
    const float* out_b  = (const float*)d_in[6];
    const float* rel    = (const float*)d_in[7];
    const float* ln2_g  = (const float*)d_in[8];
    const float* ln2_b  = (const float*)d_in[9];
    const float* ff_w1  = (const float*)d_in[10];
    const float* ff_b1  = (const float*)d_in[11];
    const float* ff_w2  = (const float*)d_in[12];
    const float* ff_b2  = (const float*)d_in[13];
    float* out = (float*)d_out;

    char* p = (char*)d_ws;
    auto carve = [&](size_t bytes) { void* r = (void*)p; p += align256(bytes); return r; };

    float*     xs     = (float*)    carve((size_t)BN_ * C_ * 4);           // residual base
    _Float16*  h16    = (_Float16*) carve((size_t)BN_ * C_ * 2);           // LN out (reused for LN2)
    _Float16*  wqkvT  = (_Float16*) carve((size_t)(3 * INNER_) * C_ * 2);  // [768,512]
    _Float16*  woutT  = (_Float16*) carve((size_t)C_ * INNER_ * 2);        // [512,256]
    _Float16*  wff1T  = (_Float16*) carve((size_t)HIDDEN_ * C_ * 2);       // [2048,512]
    _Float16*  wff2T  = (_Float16*) carve((size_t)C_ * HIDDEN_ * 2);       // [512,2048]
    _Float16*  qkv16  = (_Float16*) carve((size_t)BN_ * 3 * INNER_ * 2);
    _Float16*  q16    = (_Float16*) carve((size_t)BH_ * N_ * DHEAD_ * 2);
    _Float16*  k16    = (_Float16*) carve((size_t)BH_ * N_ * DHEAD_ * 2);
    _Float16*  vT16   = (_Float16*) carve((size_t)BH_ * DHEAD_ * NP_ * 2);
    float*     scores = (float*)    carve((size_t)BH_ * N_ * NP_ * 4);
    _Float16*  P16    = (_Float16*)scores;                                 // in-place alias (safe)
    _Float16*  o16    = (_Float16*) carve((size_t)BN_ * INNER_ * 2);
    float*     xs2    = (float*)    carve((size_t)BN_ * C_ * 4);
    _Float16*  ffh16  = (_Float16*) carve((size_t)BN_ * HIDDEN_ * 2);
    float*     xs3    = (float*)    carve((size_t)BN_ * C_ * 4);

    auto cdiv = [](size_t a, size_t b) { return (unsigned)((a + b - 1) / b); };

    // 1) Convert + transpose weights to f16 [N,K]
    wconv_kernel<<<cdiv((size_t)C_ * 3 * INNER_, 256), 256, 0, stream>>>(qkv_w, wqkvT, C_, 3 * INNER_);
    wconv_kernel<<<cdiv((size_t)INNER_ * C_, 256), 256, 0, stream>>>(out_w, woutT, INNER_, C_);
    wconv_kernel<<<cdiv((size_t)C_ * HIDDEN_, 256), 256, 0, stream>>>(ff_w1, wff1T, C_, HIDDEN_);
    wconv_kernel<<<cdiv((size_t)HIDDEN_ * C_, 256), 256, 0, stream>>>(ff_w2, wff2T, HIDDEN_, C_);

    // 2) LN1 over transposed x; emits xs (f32 residual base) and h16 (f16)
    ln_kernel<<<BN_, 256, 0, stream>>>(x, ln1_g, ln1_b, xs, h16, /*sn=*/1, /*sc=*/N_);

    // 3) QKV projection: [12544,512] @ [512,768]  (TDM-staged WMMA GEMM)
    gemm16_kernel<<<dim3(3 * INNER_ / 128, BN_ / 64), 256, 0, stream>>>(
        h16, wqkvT, qkv_b, qkv16, nullptr, nullptr, BN_, 3 * INNER_, C_, 0);

    // 4) Repack into per-head q (pre-scaled), k, and transposed/padded v
    repack_qkv_kernel<<<cdiv((size_t)BH_ * NP_ * DHEAD_, 256), 256, 0, stream>>>(qkv16, q16, k16, vT16);

    // 5) Scores + relative-position bias (1 WMMA per 16x16 tile, K=32)
    attn_scores_kernel<<<dim3(cdiv((size_t)NT_ * NT_, 8), BH_), 256, 0, stream>>>(q16, k16, rel, scores);

    // 6) Row softmax -> P (f16, padded to 800, aliases the score buffer)
    softmax_kernel<<<BH_ * N_, 256, 0, stream>>>(scores, P16);

    // 7) O = P @ V : per (b,h) [784,800] @ [800,32]
    attn_out_kernel<<<dim3(cdiv(NT_, 8), BH_), 256, 0, stream>>>(P16, vT16, o16);

    // 8) Out projection + residual: xs2 = xs + o @ out_w + out_b
    gemm16_kernel<<<dim3(C_ / 128, BN_ / 64), 256, 0, stream>>>(
        o16, woutT, out_b, nullptr, xs2, xs, BN_, C_, INNER_, 2);

    // 9) LN2 (contiguous read), reuse h16
    ln_kernel<<<BN_, 256, 0, stream>>>(xs2, ln2_g, ln2_b, nullptr, h16, /*sn=*/C_, /*sc=*/1);

    // 10) FF1 + exact GELU: [12544,512] @ [512,2048]
    gemm16_kernel<<<dim3(HIDDEN_ / 128, BN_ / 64), 256, 0, stream>>>(
        h16, wff1T, ff_b1, ffh16, nullptr, nullptr, BN_, HIDDEN_, C_, 1);

    // 11) FF2 + residual: xs3 = xs2 + ffh @ ff_w2 + ff_b2
    gemm16_kernel<<<dim3(C_ / 128, BN_ / 64), 256, 0, stream>>>(
        ffh16, wff2T, ff_b2, nullptr, xs3, xs2, BN_, C_, HIDDEN_, 2);

    // 12) [B,N,C] -> [B,C,28,28]
    out_transpose_kernel<<<cdiv((size_t)B_ * C_ * N_, 256), 256, 0, stream>>>(xs3, out);
}
